// Attn_31241592111494
// MI455X (gfx1250) — matrix operand order repaired and checked by reading, not compiled, not run on hardware
//
#include <hip/hip_runtime.h>
#include <hip/hip_bf16.h>

typedef __attribute__((ext_vector_type(16))) __bf16 v16bf;
typedef __attribute__((ext_vector_type(8)))  __bf16 v8bf;
typedef __attribute__((ext_vector_type(4)))  __bf16 v4bf;
typedef __attribute__((ext_vector_type(8)))  float  v8f;
typedef __attribute__((ext_vector_type(4)))  float  v4f;

#define HIDDEN    1024
#define STATE_LEN 8192
#define SEQ_LEN   8192

// ---------------------------------------------------------------------------
// Tiling: 256 threads = 8 wave32s; block tile 128(M) x 64(N); K-step 32.
// Waves 4(M) x 2(N); wave tile 32x32 = 2x2 WMMA accs; bf16x3 -> 12 wmma/step.
// ---------------------------------------------------------------------------
constexpr int TM = 128, TN = 64, TK = 32;
constexpr int ASTR = TK + 8;                 // 40 bf16 = 80 B row stride
constexpr int BSTR = TN + 8;                 // 72 bf16 = 144 B row stride
constexpr int A_ELEMS   = TM * ASTR;         // 5120
constexpr int B_ELEMS   = TK * BSTR;         // 2304
constexpr int BUF_ELEMS = 2 * A_ELEMS + 2 * B_ELEMS;  // 14848 bf16 = 29696 B

// Async global->LDS 16-byte copy (GV mode), tracked by ASYNCcnt.
__device__ __forceinline__ void async_b128(unsigned lds_byte, const void* gaddr) {
  asm volatile("global_load_async_to_lds_b128 %0, %1, off"
               :: "v"(lds_byte), "v"(gaddr) : "memory");
}
__device__ __forceinline__ void wait_async0() {
  asm volatile("s_wait_asynccnt 0x0" ::: "memory");
}

__device__ __forceinline__ __bf16 bf_hi(float x) { return (__bf16)x; }
__device__ __forceinline__ __bf16 bf_lo(float x, __bf16 h) { return (__bf16)(x - (float)h); }

// ---------------------------------------------------------------------------
// GEMM (operands pre-split to bf16 hi/lo):
//   C[M,N] = (Ahi+Alo)[M,K] @ (BThi+BTlo)[K,N]  (+ bias[N])
// EPI = 0: f32 row-major C.   EPI = 1: write C^T as bf16 hi/lo (CT[N][M]).
// ---------------------------------------------------------------------------
template <int EPI, bool HAS_BIAS>
__global__ __launch_bounds__(256)
void gemm_bf16x3(const __bf16* __restrict__ Ahi, const __bf16* __restrict__ Alo,
                 const __bf16* __restrict__ BThi, const __bf16* __restrict__ BTlo,
                 const float* __restrict__ bias,
                 float* __restrict__ Cf32,
                 __bf16* __restrict__ CThi, __bf16* __restrict__ CTlo,
                 int M, int N, int K) {
  __shared__ __bf16 smem[2 * BUF_ELEMS];   // assumed at LDS offset 0

  const int t    = threadIdx.x;
  const int lane = t & 31;
  const int wave = t >> 5;
  const int wm   = wave & 3;
  const int wn   = wave >> 2;
  const int Mbase = blockIdx.y * TM;
  const int Nbase = blockIdx.x * TN;

  // per-thread staging coordinates
  const int arow0 = t >> 2, aseg = (t & 3) * 8;          // A: idx t   -> row, k-seg
  const int arow1 = (t + 256) >> 2;                      //    idx t+256
  const int brow  = t >> 3, bseg = (t & 7) * 8;          // B: 32 rows x 8 segs

  auto prefetch = [&](int buf, int k0) {
    const unsigned base = (unsigned)(buf * BUF_ELEMS * 2);
    // A hi/lo: 128 rows x 64B, 2 x 16B per thread per array
    async_b128(base + (arow0 * ASTR + aseg) * 2,
               Ahi + (size_t)(Mbase + arow0) * K + k0 + aseg);
    async_b128(base + (arow1 * ASTR + aseg) * 2,
               Ahi + (size_t)(Mbase + arow1) * K + k0 + aseg);
    async_b128(base + (A_ELEMS + arow0 * ASTR + aseg) * 2,
               Alo + (size_t)(Mbase + arow0) * K + k0 + aseg);
    async_b128(base + (A_ELEMS + arow1 * ASTR + aseg) * 2,
               Alo + (size_t)(Mbase + arow1) * K + k0 + aseg);
    // B hi/lo: 32 rows x 128B, 1 x 16B per thread per array
    async_b128(base + (2 * A_ELEMS + brow * BSTR + bseg) * 2,
               BThi + (size_t)(k0 + brow) * N + Nbase + bseg);
    async_b128(base + (2 * A_ELEMS + B_ELEMS + brow * BSTR + bseg) * 2,
               BTlo + (size_t)(k0 + brow) * N + Nbase + bseg);
  };

  v8f acc[2][2] = {};

  prefetch(0, 0);
  int cur = 0;
  for (int k0 = 0; k0 < K; k0 += TK, cur ^= 1) {
    wait_async0();        // this wave's copies into `cur` are done
    __syncthreads();      // all waves' copies done; all prior reads done
    if (k0 + TK < K) prefetch(cur ^ 1, k0 + TK);

    const __bf16* As_hi = smem + cur * BUF_ELEMS;
    const __bf16* As_lo = As_hi + A_ELEMS;
    const __bf16* Bs_hi = As_hi + 2 * A_ELEMS;
    const __bf16* Bs_lo = Bs_hi + B_ELEMS;

    // A frags (ISA 16-bit A 16x32): lane<16: M=lane, K{0..7,16..23};
    //                               lane>=16: M=lane-16, K{8..15,24..31}
    const int mrow = wm * 32 + (lane & 15);
    const int koff = (lane < 16) ? 0 : 8;
    v16bf a_hi[2], a_lo[2], b_hi[2], b_lo[2];
    #pragma unroll
    for (int mi = 0; mi < 2; ++mi) {
      const __bf16* ph = As_hi + (mrow + mi * 16) * ASTR + koff;
      const __bf16* pl = As_lo + (mrow + mi * 16) * ASTR + koff;
      v8bf h0 = *(const v8bf*)ph, h1 = *(const v8bf*)(ph + 16);
      v8bf l0 = *(const v8bf*)pl, l1 = *(const v8bf*)(pl + 16);
      #pragma unroll
      for (int e = 0; e < 8; ++e) {
        a_hi[mi][e] = h0[e]; a_hi[mi][8 + e] = h1[e];
        a_lo[mi][e] = l0[e]; a_lo[mi][8 + e] = l1[e];
      }
    }
    // B frags (ISA B 32x16): N = lane&15, K same order as A (koff+{0..7}, 16+koff+{0..7})
    #pragma unroll
    for (int ni = 0; ni < 2; ++ni) {
      const __bf16* ph = Bs_hi + koff * BSTR + wn * 32 + ni * 16 + (lane & 15);
      const __bf16* pl = Bs_lo + koff * BSTR + wn * 32 + ni * 16 + (lane & 15);
      #pragma unroll
      for (int e = 0; e < 8; ++e) {
        b_hi[ni][e] = ph[e * BSTR]; b_hi[ni][8 + e] = ph[(16 + e) * BSTR];
        b_lo[ni][e] = pl[e * BSTR]; b_lo[ni][8 + e] = pl[(16 + e) * BSTR];
      }
    }

    #pragma unroll
    for (int mi = 0; mi < 2; ++mi)
      #pragma unroll
      for (int ni = 0; ni < 2; ++ni) {
        acc[mi][ni] = __builtin_amdgcn_wmma_f32_16x16x32_bf16(
            false, a_hi[mi], false, b_hi[ni], (short)0, acc[mi][ni], false, false);
        acc[mi][ni] = __builtin_amdgcn_wmma_f32_16x16x32_bf16(
            false, a_hi[mi], false, b_lo[ni], (short)0, acc[mi][ni], false, false);
        acc[mi][ni] = __builtin_amdgcn_wmma_f32_16x16x32_bf16(
            false, a_lo[mi], false, b_hi[ni], (short)0, acc[mi][ni], false, false);
      }
  }

  // Epilogue. ISA C layout: VGPR r -> M = r + 8*(lane>=16), N = lane&15.
  const int mtop = Mbase + wm * 32 + ((lane >> 4) << 3);
  const int ncol = Nbase + wn * 32 + (lane & 15);
  #pragma unroll
  for (int ni = 0; ni < 2; ++ni) {
    const int col = ncol + ni * 16;
    const float bv = HAS_BIAS ? bias[col] : 0.0f;
    #pragma unroll
    for (int mi = 0; mi < 2; ++mi)
      #pragma unroll
      for (int r = 0; r < 8; ++r) {
        const int row = mtop + mi * 16 + r;
        const float val = acc[mi][ni][r] + bv;
        if (EPI == 0) {
          Cf32[(size_t)row * N + col] = val;
        } else {
          const __bf16 h = bf_hi(val);
          CThi[(size_t)col * M + row] = h;
          CTlo[(size_t)col * M + row] = bf_lo(val, h);
        }
      }
  }
}

// ---------------------------------------------------------------------------
// One-time f32 -> bf16 hi/lo split (layout-preserving). 4 elems/thread.
// ---------------------------------------------------------------------------
__global__ __launch_bounds__(256)
void split_f32(const float* __restrict__ in, __bf16* __restrict__ hi,
               __bf16* __restrict__ lo) {
  const size_t i = ((size_t)blockIdx.x * 256 + threadIdx.x) * 4;
  v4f v = *(const v4f*)(in + i);
  v4bf h, l;
  #pragma unroll
  for (int e = 0; e < 4; ++e) {
    h[e] = bf_hi(v[e]);
    l[e] = bf_lo(v[e], h[e]);
  }
  *(v4bf*)(hi + i) = h;
  *(v4bf*)(lo + i) = l;
}

// f32 [R][Ccols] -> transposed bf16 hi/lo [Ccols][R]
__global__ __launch_bounds__(256)
void split_T_f32(const float* __restrict__ in, __bf16* __restrict__ thi,
                 __bf16* __restrict__ tlo, int R, int Ccols) {
  const size_t idx = (size_t)blockIdx.x * 256 + threadIdx.x;
  const int r = (int)(idx / Ccols), c = (int)(idx % Ccols);
  const float x = in[idx];
  const __bf16 h = bf_hi(x);
  thi[(size_t)c * R + r] = h;
  tlo[(size_t)c * R + r] = bf_lo(x, h);
}

// ---------------------------------------------------------------------------
// In-place row softmax over 8192-wide rows; 32 values/thread in registers.
// ---------------------------------------------------------------------------
__global__ __launch_bounds__(256)
void softmax_rows(float* __restrict__ E) {
  float* rp = E + (size_t)blockIdx.x * SEQ_LEN;
  const int t = threadIdx.x;

  float v[32];
  float m = -3.402823466e38f;
  #pragma unroll
  for (int i = 0; i < 32; ++i) { v[i] = rp[t + i * 256]; m = fmaxf(m, v[i]); }
  #pragma unroll
  for (int off = 16; off > 0; off >>= 1) m = fmaxf(m, __shfl_xor(m, off, 32));

  __shared__ float redm[8];
  if ((t & 31) == 0) redm[t >> 5] = m;
  __syncthreads();
  m = redm[0];
  #pragma unroll
  for (int w = 1; w < 8; ++w) m = fmaxf(m, redm[w]);

  float s = 0.0f;
  #pragma unroll
  for (int i = 0; i < 32; ++i) { v[i] = __expf(v[i] - m); s += v[i]; }
  #pragma unroll
  for (int off = 16; off > 0; off >>= 1) s += __shfl_xor(s, off, 32);

  __shared__ float reds[8];
  if ((t & 31) == 0) reds[t >> 5] = s;
  __syncthreads();
  s = 0.0f;
  #pragma unroll
  for (int w = 0; w < 8; ++w) s += reds[w];

  const float inv = 1.0f / s;
  #pragma unroll
  for (int i = 0; i < 32; ++i) rp[t + i * 256] = v[i] * inv;
}

// ---------------------------------------------------------------------------
extern "C" void kernel_launch(void* const* d_in, const int* in_sizes, int n_in,
                              void* d_out, int out_size, void* d_ws, size_t ws_size,
                              hipStream_t stream) {
  const float* out_state = (const float*)d_in[0];   // [8192,1024]
  const float* history   = (const float*)d_in[1];   // [8192,1024]
  const float* attn_W    = (const float*)d_in[2];   // [1024,1024]
  const float* attn_b    = (const float*)d_in[3];   // [1024]
  float* out = (float*)d_out;                       // [8192,8192]

  // workspace layout (bytes); total = 100 MB
  char* w = (char*)d_ws;
  const size_t MB = 1024ull * 1024ull;
  __bf16* os_hi = (__bf16*)(w + 0 * MB);    // out_state hi  [8192,1024]
  __bf16* os_lo = (__bf16*)(w + 16 * MB);
  __bf16* h_hi  = (__bf16*)(w + 32 * MB);   // history hi    [8192,1024]
  __bf16* h_lo  = (__bf16*)(w + 48 * MB);
  __bf16* wt_hi = (__bf16*)(w + 64 * MB);   // W^T hi        [1024,1024]
  __bf16* wt_lo = (__bf16*)(w + 66 * MB);
  __bf16* pt_hi = (__bf16*)(w + 68 * MB);   // proj^T hi     [1024,8192]
  __bf16* pt_lo = (__bf16*)(w + 84 * MB);

  dim3 blk(256);

  // one-time splits (elementwise, memory bound)
  split_f32<<<STATE_LEN * HIDDEN / 1024, blk, 0, stream>>>(out_state, os_hi, os_lo);
  split_f32<<<SEQ_LEN * HIDDEN / 1024, blk, 0, stream>>>(history, h_hi, h_lo);
  split_T_f32<<<HIDDEN * HIDDEN / 256, blk, 0, stream>>>(attn_W, wt_hi, wt_lo,
                                                         HIDDEN, HIDDEN);

  // proj^T = (history @ W^T + b)^T, emitted directly as bf16 hi/lo
  dim3 g1(HIDDEN / TN, SEQ_LEN / TM);               // (16, 64)
  gemm_bf16x3<1, true><<<g1, blk, 0, stream>>>(h_hi, h_lo, wt_hi, wt_lo, attn_b,
                                               nullptr, pt_hi, pt_lo,
                                               SEQ_LEN, HIDDEN, HIDDEN);

  // energies = out_state @ proj^T -> d_out (f32)
  dim3 g2(SEQ_LEN / TN, STATE_LEN / TM);            // (128, 64)
  gemm_bf16x3<0, false><<<g2, blk, 0, stream>>>(os_hi, os_lo, pt_hi, pt_lo, nullptr,
                                                out, nullptr, nullptr,
                                                STATE_LEN, SEQ_LEN, HIDDEN);

  // softmax over rows, in place
  softmax_rows<<<STATE_LEN, blk, 0, stream>>>(out);
}
